// MultiHeadAttention_23261542875370
// MI455X (gfx1250) — compile-verified, matrix operand order checked by reading, not run on hardware
//
#include <hip/hip_runtime.h>

// ---------------------------------------------------------------------------
// MHA for MI455X (gfx1250, wave32, WMMA). bf16 matrix inputs, f32 accumulate.
// - Weights pre-transposed once -> all GEMM tile fills are async HBM->LDS b128
// - V stored transposed by the QKV GEMM -> attention fills are async too
// - Double-buffered LDS: async DMA for tile k+1 overlaps WMMA on tile k
// - Flash attention, KV chunks of 64, DPP16 butterfly reductions
// ---------------------------------------------------------------------------

typedef __bf16 bf16;
typedef __bf16 v4bf  __attribute__((ext_vector_type(4)));
typedef __bf16 v8bf  __attribute__((ext_vector_type(8)));
typedef __bf16 v16bf __attribute__((ext_vector_type(16)));
typedef float  v8f   __attribute__((ext_vector_type(8)));
typedef int    v4i   __attribute__((ext_vector_type(4)));

#define EMB_  1024
#define NSEQ  2048
#define BATCH 2
#define HEADS 16
#define HDIM  64

#if defined(__gfx1250__) && __has_builtin(__builtin_amdgcn_global_load_async_to_lds_b128) && __has_builtin(__builtin_amdgcn_s_wait_asynccnt)
#define HAVE_ASYNC_LDS 1
#else
#define HAVE_ASYNC_LDS 0
#endif

__device__ __forceinline__ bf16 f2bf(float f) { return (bf16)f; }

// ---- DPP16 butterfly reductions over the 16-lane half-wave ---------------
template <int CTRL>
__device__ __forceinline__ float dppf(float v) {
  return __builtin_bit_cast(
      float, __builtin_amdgcn_mov_dpp(__builtin_bit_cast(int, v), CTRL, 0xf, 0xf, true));
}
__device__ __forceinline__ float redmax16(float v) {
  v = fmaxf(v, dppf<0xB1>(v));    // quad_perm [1,0,3,2]
  v = fmaxf(v, dppf<0x4E>(v));    // quad_perm [2,3,0,1]
  v = fmaxf(v, dppf<0x141>(v));   // row_half_mirror
  v = fmaxf(v, dppf<0x140>(v));   // row_mirror
  return v;
}
__device__ __forceinline__ float redsum16(float v) {
  v += dppf<0xB1>(v);
  v += dppf<0x4E>(v);
  v += dppf<0x141>(v);
  v += dppf<0x140>(v);
  return v;
}

// Per-lane 16-bit operand fragment: elements 0..7 = K+{0..7},
// elements 8..15 = K+16+{0..7}; K already includes (lane>=16)*8.
__device__ __forceinline__ v16bf load_frag(const bf16* base) {
  v8bf lo = *(const v8bf*)(base);
  v8bf hi = *(const v8bf*)(base + 16);
  return __builtin_shufflevector(lo, hi, 0, 1, 2, 3, 4, 5, 6, 7,
                                         8, 9, 10, 11, 12, 13, 14, 15);
}

// ---------------------------------------------------------------------------
// fp32 -> bf16 bulk convert (row-major, for x)
// ---------------------------------------------------------------------------
__global__ __launch_bounds__(256) void cvt_f32_bf16(const float* __restrict__ in,
                                                    bf16* __restrict__ out, int n4) {
  int i = blockIdx.x * 256 + threadIdx.x;
  if (i >= n4) return;
  float4 v = ((const float4*)in)[i];
  v4bf o;
  o[0] = f2bf(v.x); o[1] = f2bf(v.y); o[2] = f2bf(v.z); o[3] = f2bf(v.w);
  ((v4bf*)out)[i] = o;
}

// ---------------------------------------------------------------------------
// fp32 W[K][N] -> bf16 W^T[N][K], 32x32 LDS tile transpose
// ---------------------------------------------------------------------------
__global__ __launch_bounds__(256) void cvt_transpose_f32_bf16(
    const float* __restrict__ in, bf16* __restrict__ out, int K, int N) {
  __shared__ float tile[32][33];
  const int tx = threadIdx.x;        // 0..31
  const int ty = threadIdx.y;        // 0..7
  const int n0 = blockIdx.x * 32;
  const int k0 = blockIdx.y * 32;
#pragma unroll
  for (int j = 0; j < 4; ++j)
    tile[ty + j * 8][tx] = in[(size_t)(k0 + ty + j * 8) * N + n0 + tx];
  __syncthreads();
#pragma unroll
  for (int j = 0; j < 4; ++j)
    out[(size_t)(n0 + ty + j * 8) * K + k0 + tx] = f2bf(tile[tx][ty + j * 8]);
}

// ---------------------------------------------------------------------------
// Tiled bf16 GEMM (B pre-transposed), f32 accumulate:
//   C[M,N] = A[M,K] @ B^T[N,K]^T + bias[N]
// Block tile 128x128, 8 waves (4M x 2N), wave tile 32x64, K-step 64,
// double-buffered async LDS fills.
// MODE 0: fp32 store.  MODE 1: de-interleave (h,d,3) -> Q (pre-scaled), K,
//         V stored transposed [b,h][d][n].
// ---------------------------------------------------------------------------
#define GT_M 128
#define GT_N 128
#define GT_K 64
#define LDS_W 72   // halfword stride: 16B aligned, conflict-free frag loads

template <int MODE>
__global__ __launch_bounds__(256) void gemm_bf16_kernel(
    const bf16* __restrict__ A, const bf16* __restrict__ BT,
    const float* __restrict__ bias,
    float* __restrict__ Cf,
    bf16* __restrict__ Qo, bf16* __restrict__ Ko, bf16* __restrict__ Vo,
    int M, int N, int K) {
  __shared__ bf16 As[2][GT_M * LDS_W];
  __shared__ bf16 Bs[2][GT_N * LDS_W];   // [n][k]

  const int tid  = threadIdx.x;
  const int lane = tid & 31;
  const int wid  = tid >> 5;
  const int lr   = lane & 15;
  const int lh   = lane >> 4;
  const int m0   = blockIdx.y * GT_M;
  const int n0   = blockIdx.x * GT_N;
  const int mw   = (wid & 3) * 32;
  const int nw   = (wid >> 2) * 64;

  v8f acc[2][4];
#pragma unroll
  for (int mi = 0; mi < 2; ++mi)
#pragma unroll
    for (int ni = 0; ni < 4; ++ni)
#pragma unroll
      for (int i = 0; i < 8; ++i) acc[mi][ni][i] = 0.f;

  const int frow = tid >> 1;          // 0..127  (A row / B^T row)
  const int fcol = (tid & 1) * 32;    // 0,32

  auto fill = [&](int buf, int kt) {
    const bf16* gA = A  + (size_t)(m0 + frow) * K + kt + fcol;
    const bf16* gB = BT + (size_t)(n0 + frow) * K + kt + fcol;
#if HAVE_ASYNC_LDS
#pragma unroll
    for (int j = 0; j < 4; ++j) {
      __builtin_amdgcn_global_load_async_to_lds_b128(
          (v4i*)(gA + j * 8), (v4i*)(&As[buf][frow * LDS_W + fcol + j * 8]), 0, 0);
      __builtin_amdgcn_global_load_async_to_lds_b128(
          (v4i*)(gB + j * 8), (v4i*)(&Bs[buf][frow * LDS_W + fcol + j * 8]), 0, 0);
    }
#else
#pragma unroll
    for (int j = 0; j < 4; ++j) {
      *(v8bf*)(&As[buf][frow * LDS_W + fcol + j * 8]) = *(const v8bf*)(gA + j * 8);
      *(v8bf*)(&Bs[buf][frow * LDS_W + fcol + j * 8]) = *(const v8bf*)(gB + j * 8);
    }
#endif
  };

  fill(0, 0);
#if HAVE_ASYNC_LDS
  __builtin_amdgcn_s_wait_asynccnt(0);
#endif
  __syncthreads();

  int cur = 0;
  for (int kt = 0; kt < K; kt += GT_K) {
    const int nxt = cur ^ 1;
    if (kt + GT_K < K) fill(nxt, kt + GT_K);   // DMA next tile while computing

#pragma unroll
    for (int sl = 0; sl < 2; ++sl) {
      v16bf af[2], bfr[4];
#pragma unroll
      for (int mi = 0; mi < 2; ++mi)
        af[mi] = load_frag(&As[cur][(mw + mi * 16 + lr) * LDS_W + sl * 32 + lh * 8]);
#pragma unroll
      for (int ni = 0; ni < 4; ++ni)
        bfr[ni] = load_frag(&Bs[cur][(nw + ni * 16 + lr) * LDS_W + sl * 32 + lh * 8]);
#pragma unroll
      for (int mi = 0; mi < 2; ++mi)
#pragma unroll
        for (int ni = 0; ni < 4; ++ni)
          acc[mi][ni] = __builtin_amdgcn_wmma_f32_16x16x32_bf16(
              false, af[mi], false, bfr[ni], (short)0, acc[mi][ni], false, false);
    }

#if HAVE_ASYNC_LDS
    __builtin_amdgcn_s_wait_asynccnt(0);
#endif
    __syncthreads();
    cur = nxt;
  }

#pragma unroll
  for (int mi = 0; mi < 2; ++mi) {
#pragma unroll
    for (int ni = 0; ni < 4; ++ni) {
      const int col = n0 + nw + ni * 16 + lr;
      const float bv = bias[col];
      if (MODE == 0) {
#pragma unroll
        for (int i = 0; i < 8; ++i) {
          const int row = m0 + mw + mi * 16 + lh * 8 + i;
          Cf[(size_t)row * N + col] = acc[mi][ni][i] + bv;
        }
      } else {
        // qkv column order is (h, d, which): col = h*192 + d*3 + which
        const int h   = col / 192;
        const int rem = col - h * 192;
        const int dd  = rem / 3;
        const int t   = rem - dd * 3;
        const float mulf = (t == 0) ? 0.03125f : 1.0f;  // 1/sqrt(1024) into Q
#pragma unroll
        for (int i = 0; i < 8; ++i) {
          const int row = m0 + mw + mi * 16 + lh * 8 + i;
          const int bb = row >> 11;
          const int nn = row & 2047;
          const bf16 val = f2bf((acc[mi][ni][i] + bv) * mulf);
          if (t == 2) {
            // V stored transposed: [b,h][d][n]
            Vo[((size_t)(bb * HEADS + h) * HDIM + dd) * NSEQ + nn] = val;
          } else {
            bf16* dst = (t == 0) ? Qo : Ko;
            dst[((size_t)(bb * HEADS + h) * NSEQ + nn) * HDIM + dd] = val;
          }
        }
      }
    }
  }
}

// ---------------------------------------------------------------------------
// Flash attention: per (b,h), 4 waves/block, 16 q-rows per wave, KV chunks of
// 64 (4 S col-groups reduced together), double-buffered async K/V fills.
// Q pre-scaled; V arrives transposed. ctx[b, n, h*64+d] bf16 output.
// ---------------------------------------------------------------------------
#define KVC   64

__global__ __launch_bounds__(128) void attn_kernel(
    const bf16* __restrict__ Q, const bf16* __restrict__ K,
    const bf16* __restrict__ V, bf16* __restrict__ ctx) {
  __shared__ bf16 Ks[2][KVC * LDS_W];    // K chunk, [kv][d]
  __shared__ bf16 Vs[2][HDIM * LDS_W];   // V chunk, [d][kv] (global transposed)
  __shared__ bf16 Ps[4][16 * LDS_W];     // per-wave P tile [q][kv]

  const int tid  = threadIdx.x;
  const int lane = tid & 31;
  const int wid  = tid >> 5;
  const int lr   = lane & 15;
  const int lh   = lane >> 4;
  const int bh   = blockIdx.y;
  const int b    = bh >> 4;
  const int h    = bh & 15;
  const int q0   = blockIdx.x * 64 + wid * 16;

  const size_t base = (size_t)bh * NSEQ * HDIM;
  const bf16* Qp = Q + base;
  const bf16* Kp = K + base;
  const bf16* Vp = V + base;              // [d][n] layout

  v16bf qa[2];
#pragma unroll
  for (int sl = 0; sl < 2; ++sl)
    qa[sl] = load_frag(Qp + (size_t)(q0 + lr) * HDIM + sl * 32 + lh * 8);

  float rowm[8], rowl[8];
  v8f o[4];
#pragma unroll
  for (int i = 0; i < 8; ++i) { rowm[i] = -INFINITY; rowl[i] = 0.f; }
#pragma unroll
  for (int g = 0; g < 4; ++g)
#pragma unroll
    for (int i = 0; i < 8; ++i) o[g][i] = 0.f;

  const int frow = tid >> 1;          // 0..63 (K: kv row, V: d row)
  const int fcol = (tid & 1) * 32;    // 0,32

  auto fill = [&](int buf, int kv) {
    const bf16* gK = Kp + (size_t)(kv + frow) * HDIM + fcol;
    const bf16* gV = Vp + (size_t)frow * NSEQ + kv + fcol;
#if HAVE_ASYNC_LDS
#pragma unroll
    for (int j = 0; j < 4; ++j) {
      __builtin_amdgcn_global_load_async_to_lds_b128(
          (v4i*)(gK + j * 8), (v4i*)(&Ks[buf][frow * LDS_W + fcol + j * 8]), 0, 0);
      __builtin_amdgcn_global_load_async_to_lds_b128(
          (v4i*)(gV + j * 8), (v4i*)(&Vs[buf][frow * LDS_W + fcol + j * 8]), 0, 0);
    }
#else
#pragma unroll
    for (int j = 0; j < 4; ++j) {
      *(v8bf*)(&Ks[buf][frow * LDS_W + fcol + j * 8]) = *(const v8bf*)(gK + j * 8);
      *(v8bf*)(&Vs[buf][frow * LDS_W + fcol + j * 8]) = *(const v8bf*)(gV + j * 8);
    }
#endif
  };

  fill(0, 0);
#if HAVE_ASYNC_LDS
  __builtin_amdgcn_s_wait_asynccnt(0);
#endif
  __syncthreads();

  int cur = 0;
  for (int kv = 0; kv < NSEQ; kv += KVC) {
    const int nxt = cur ^ 1;
    if (kv + KVC < NSEQ) fill(nxt, kv + KVC);  // DMA next chunk during compute

    // S = Q @ K^T : 16 x 64, four 16-col groups, K-dim 64 = 2 WMMA each
    v8f s[4];
#pragma unroll
    for (int g = 0; g < 4; ++g) {
#pragma unroll
      for (int i = 0; i < 8; ++i) s[g][i] = 0.f;
#pragma unroll
      for (int sl = 0; sl < 2; ++sl) {
        v16bf kb = load_frag(&Ks[cur][(g * 16 + lr) * LDS_W + sl * 32 + lh * 8]);
        s[g] = __builtin_amdgcn_wmma_f32_16x16x32_bf16(
            false, qa[sl], false, kb, (short)0, s[g], false, false);
      }
    }

    // online softmax; row r = i + lh*8, its 16 cols sit across the half-wave
    bf16* pw = &Ps[wid][0];
#pragma unroll
    for (int i = 0; i < 8; ++i) {
      const float cm = redmax16(fmaxf(fmaxf(s[0][i], s[1][i]),
                                      fmaxf(s[2][i], s[3][i])));
      const float newm  = fmaxf(rowm[i], cm);
      const float alpha = __expf(rowm[i] - newm);
      rowm[i] = newm;
      float p[4];
#pragma unroll
      for (int g = 0; g < 4; ++g) p[g] = __expf(s[g][i] - newm);
      const float rs = redsum16((p[0] + p[1]) + (p[2] + p[3]));
      rowl[i] = rowl[i] * alpha + rs;
#pragma unroll
      for (int g = 0; g < 4; ++g) o[g][i] *= alpha;
#pragma unroll
      for (int g = 0; g < 4; ++g)
        pw[(lh * 8 + i) * LDS_W + g * 16 + lr] = f2bf(p[g]);
    }
    asm volatile("" ::: "memory");  // keep P store -> P frag load ordered

    // O += P @ V  (P: 16x64 A-operand read back in A-layout; V^T in LDS)
    v16bf pa[2];
#pragma unroll
    for (int sl = 0; sl < 2; ++sl)
      pa[sl] = load_frag(pw + lr * LDS_W + sl * 32 + lh * 8);
#pragma unroll
    for (int g = 0; g < 4; ++g) {
#pragma unroll
      for (int sl = 0; sl < 2; ++sl) {
        v16bf vb = load_frag(&Vs[cur][(g * 16 + lr) * LDS_W + sl * 32 + lh * 8]);
        o[g] = __builtin_amdgcn_wmma_f32_16x16x32_bf16(
            false, pa[sl], false, vb, (short)0, o[g], false, false);
      }
    }

#if HAVE_ASYNC_LDS
    __builtin_amdgcn_s_wait_asynccnt(0);
#endif
    __syncthreads();
    cur = nxt;
  }

#pragma unroll
  for (int i = 0; i < 8; ++i) rowl[i] = 1.0f / rowl[i];
  const int row0 = b * NSEQ + q0;
#pragma unroll
  for (int g = 0; g < 4; ++g) {
    const int col = h * HDIM + g * 16 + lr;
#pragma unroll
    for (int i = 0; i < 8; ++i) {
      const int row = row0 + lh * 8 + i;
      ctx[(size_t)row * EMB_ + col] = f2bf(o[g][i] * rowl[i]);
    }
  }
}

// ---------------------------------------------------------------------------
// Host launcher
// ---------------------------------------------------------------------------
extern "C" void kernel_launch(void* const* d_in, const int* in_sizes, int n_in,
                              void* d_out, int out_size, void* d_ws, size_t ws_size,
                              hipStream_t stream) {
  (void)in_sizes; (void)n_in; (void)out_size; (void)ws_size;
  const float* x     = (const float*)d_in[0];
  const float* Wqkv  = (const float*)d_in[1];
  const float* bqkv  = (const float*)d_in[2];
  const float* Wproj = (const float*)d_in[3];
  const float* bproj = (const float*)d_in[4];
  float* out = (float*)d_out;

  const size_t M   = (size_t)BATCH * NSEQ;          // 4096
  const size_t N1  = 3 * EMB_;                      // 3072
  const size_t Kd  = EMB_;                          // 1024
  const size_t nX  = M * Kd;
  const size_t nW1 = Kd * N1;
  const size_t nW2 = Kd * EMB_;
  const size_t nHD = (size_t)BATCH * HEADS * NSEQ * HDIM;

  char* ws = (char*)d_ws;
  size_t off = 0;
  auto take = [&](size_t bytes) {
    size_t o = off;
    off += (bytes + 255) & ~(size_t)255;
    return o;
  };
  bf16* Xb   = (bf16*)(ws + take(nX  * 2));
  bf16* W1bT = (bf16*)(ws + take(nW1 * 2));   // [3072][1024]
  bf16* W2bT = (bf16*)(ws + take(nW2 * 2));   // [1024][1024]
  bf16* Qb   = (bf16*)(ws + take(nHD * 2));   // [b,h][n][d], pre-scaled
  bf16* Kb   = (bf16*)(ws + take(nHD * 2));   // [b,h][n][d]
  bf16* Vb   = (bf16*)(ws + take(nHD * 2));   // [b,h][d][n]  (transposed)
  bf16* Ctx  = (bf16*)(ws + take(nX  * 2));

  cvt_f32_bf16<<<(int)(nX / 4 / 256), 256, 0, stream>>>(x, Xb, (int)(nX / 4));
  cvt_transpose_f32_bf16<<<dim3((unsigned)(N1 / 32), (unsigned)(Kd / 32)),
                           dim3(32, 8), 0, stream>>>(Wqkv, W1bT, (int)Kd, (int)N1);
  cvt_transpose_f32_bf16<<<dim3((unsigned)(EMB_ / 32), (unsigned)(Kd / 32)),
                           dim3(32, 8), 0, stream>>>(Wproj, W2bT, (int)Kd, EMB_);

  // qkv projection + de-interleave into Q (scaled) / K / V^T bf16
  gemm_bf16_kernel<1><<<dim3((unsigned)(N1 / GT_N), (unsigned)(M / GT_M)), 256, 0, stream>>>(
      Xb, W1bT, bqkv, nullptr, Qb, Kb, Vb, (int)M, (int)N1, (int)Kd);

  // flash attention -> ctx [b, n, (h d)] bf16
  attn_kernel<<<dim3(NSEQ / 64, BATCH * HEADS), 128, 0, stream>>>(Qb, Kb, Vb, Ctx);

  // output projection, fp32 store
  gemm_bf16_kernel<0><<<dim3(EMB_ / GT_N, (unsigned)(M / GT_M)), 256, 0, stream>>>(
      Ctx, W2bT, bproj, out, nullptr, nullptr, nullptr, (int)M, EMB_, (int)Kd);
}
